// SMoE_36661840839480
// MI455X (gfx1250) — matrix operand order repaired and checked by reading, not compiled, not run on hardware
//
#include <hip/hip_runtime.h>

// Problem constants (from reference)
#define BB   4
#define TT   2048
#define DD   1024
#define OO   1024
#define EE   8
#define NTOK (BB*TT)      // 8192 tokens
#define NMAX NTOK         // worst-case tokens per expert

// GEMM tiling
#define TM 128
#define TN 128
#define TK 32
#define BSTRIDE 40        // padded LDS row stride (bf16 elems) -> conflict-free b128 reads
#define BBYTES (TN * BSTRIDE * 2)   // one B buffer = 10240 bytes

typedef __attribute__((ext_vector_type(16))) __bf16 v16bf;
typedef __attribute__((ext_vector_type(8)))  float  v8f;

union Pack32 { uint4 q[2]; v16bf v; };

__device__ __forceinline__ unsigned short f2bf(float f) {
  union { float f; unsigned u; } v; v.f = f;
  unsigned u = v.u;
  unsigned r = u + 0x7FFFu + ((u >> 16) & 1u);   // round-to-nearest-even
  return (unsigned short)(r >> 16);
}

// Issue two async 16B global->LDS copies (32 bytes per thread) for the B tile.
// gfx1250: tracked by ASYNCcnt; offset applies to both LDS and memory address.
__device__ __forceinline__ void async_fill32(unsigned lds_dst, const void* gsrc) {
  asm volatile("global_load_async_to_lds_b128 %0, %1, off\n\t"
               "global_load_async_to_lds_b128 %0, %1, off offset:16"
               :: "v"(lds_dst), "v"(gsrc) : "memory");
}
__device__ __forceinline__ void wait_async0() {
  asm volatile("s_wait_asynccnt 0x0" ::: "memory");
}

// ---------------- Kernel 0: zero output + per-expert counts ----------------
__global__ void k_init(float* __restrict__ out, int n, int* __restrict__ counts) {
  int i = blockIdx.x * blockDim.x + threadIdx.x;
  if (i < EE) counts[i] = 0;
  for (; i < n; i += gridDim.x * blockDim.x) out[i] = 0.f;
}

// ---------------- Kernel 1: x fp32 -> bf16 ----------------
__global__ void k_cvt_x(const float* __restrict__ x, unsigned short* __restrict__ xb, int n8) {
  int i = blockIdx.x * blockDim.x + threadIdx.x;
  if (i >= n8) return;
  const float4* src = (const float4*)x + (size_t)i * 2;
  float4 a = src[0], b = src[1];
  union { unsigned short s[8]; uint4 q; } o;
  o.s[0] = f2bf(a.x); o.s[1] = f2bf(a.y); o.s[2] = f2bf(a.z); o.s[3] = f2bf(a.w);
  o.s[4] = f2bf(b.x); o.s[5] = f2bf(b.y); o.s[6] = f2bf(b.z); o.s[7] = f2bf(b.w);
  *((uint4*)(xb) + i) = o.q;
}

// ---------------- Kernel 2: expert_w [E][D][O] fp32 -> wt [E][O][D] bf16 ----------------
__global__ void k_wt(const float* __restrict__ w, unsigned short* __restrict__ wt) {
  int i = blockIdx.x * blockDim.x + threadIdx.x;   // E * O * (D/16) = 524288
  if (i >= EE * OO * (DD / 16)) return;
  int dblk = i & 63;                 // D/16 = 64
  int o    = (i >> 6) & (OO - 1);
  int e    = i >> 16;
  int d0   = dblk * 16;
  union { unsigned short s[16]; uint4 q[2]; } buf;
  const float* src = w + ((size_t)e * DD + d0) * OO + o;
  #pragma unroll
  for (int j = 0; j < 16; ++j) buf.s[j] = f2bf(src[(size_t)j * OO]);
  uint4* dst = (uint4*)(wt + ((size_t)e * OO + o) * DD + d0);
  dst[0] = buf.q[0]; dst[1] = buf.q[1];
}

// ---------------- Kernel 3: gating + top-2 routing (wave per token) ----------------
__global__ void k_gate(const float* __restrict__ x, const float* __restrict__ gw,
                       const float* __restrict__ gb, int* __restrict__ counts,
                       int* __restrict__ tok_list, float* __restrict__ wgt_list) {
  int wave = threadIdx.x >> 5, lane = threadIdx.x & 31;
  int t = blockIdx.x * 8 + wave;
  float s[EE];
  #pragma unroll
  for (int e = 0; e < EE; ++e) s[e] = 0.f;
  const float* xr = x + (size_t)t * DD;
  for (int d = lane; d < DD; d += 32) {
    float xv = xr[d];
    const float* g = gw + (size_t)d * EE;
    #pragma unroll
    for (int e = 0; e < EE; ++e) s[e] += xv * g[e];
  }
  #pragma unroll
  for (int e = 0; e < EE; ++e) {
    #pragma unroll
    for (int off = 16; off >= 1; off >>= 1)
      s[e] += __shfl_xor(s[e], off, 32);
  }
  if (lane == 0) {
    #pragma unroll
    for (int e = 0; e < EE; ++e) s[e] += gb[e];
    int i0 = 0; float b0 = s[0];
    #pragma unroll
    for (int e = 1; e < EE; ++e) if (s[e] > b0) { b0 = s[e]; i0 = e; }  // first max
    int i1 = (i0 == 0) ? 1 : 0; float b1 = s[i1];
    #pragma unroll
    for (int e = 0; e < EE; ++e)
      if (e != i0 && s[e] > b1) { b1 = s[e]; i1 = e; }                  // first 2nd max
    float e1 = __expf(b1 - b0);
    float w0 = 1.f / (1.f + e1);
    float w1 = e1 * w0;
    int p0 = atomicAdd(&counts[i0], 1);
    tok_list[i0 * NMAX + p0] = t; wgt_list[i0 * NMAX + p0] = w0;
    int p1 = atomicAdd(&counts[i1], 1);
    tok_list[i1 * NMAX + p1] = t; wgt_list[i1 * NMAX + p1] = w1;
  }
}

// ---------------- Kernel 4: per-expert gathered GEMM with bf16 WMMA ----------------
// Block: 256 threads (8 waves). Tile: TM=128 gathered tokens x TN=128 output cols.
// B tile double-buffered in LDS via async global->LDS copies; per chunk, all 8
// B-operand register groups are preloaded (sched_barrier-enforced clause, one
// LDS-latency exposure), then the 8 WMMAs issue back-to-back. A registers are
// pipelined one chunk ahead.
__global__ void __launch_bounds__(256) k_moe_gemm(
    const unsigned short* __restrict__ xb,   // [NTOK][D] bf16
    const unsigned short* __restrict__ wt,   // [E][O][D] bf16 (pre-transposed)
    const float* __restrict__ eb,            // [E][O] bias
    const int* __restrict__ counts,
    const int* __restrict__ tok_list,
    const float* __restrict__ wgt_list,
    float* __restrict__ out) {
  __shared__ unsigned short sB[2][TN * BSTRIDE];  // 2 x 10 KB

  int e   = blockIdx.z;
  int cnt = counts[e];
  int m0  = blockIdx.y * TM;
  if (m0 >= cnt) return;                       // uniform early exit
  int n0  = blockIdx.x * TN;

  int wave = threadIdx.x >> 5, lane = threadIdx.x & 31;
  int g = lane >> 4, mr = lane & 15;
  int eoff = e * NMAX;

  // A source row for this lane (ISA A layout: row = lane%16, K halves split by lane/16)
  int slotA = m0 + wave * 16 + mr;
  int slA   = (slotA < cnt) ? slotA : (cnt - 1);     // clamp: keeps EXEC all-ones
  const unsigned short* xrow = xb + (size_t)tok_list[eoff + slA] * DD;

  // B-tile async fill addressing: LDS row = out col (n0+fc), 32 K elems per chunk
  int fc = threadIdx.x >> 1, fh = threadIdx.x & 1;
  const unsigned short* wrow = wt + ((size_t)e * OO + n0 + fc) * DD + fh * 16;
  unsigned sdst0 = (unsigned)(uintptr_t)&sB[0][fc * BSTRIDE + fh * 16];

  v8f acc[8] = {};

  // Prologue: async fill of buffer 0 (chunk 0), A registers for chunk 0.
  async_fill32(sdst0, wrow);
  Pack32 Acur, Anxt;
  Acur.q[0] = *(const uint4*)(xrow + g * 8);
  Acur.q[1] = *(const uint4*)(xrow + 16 + g * 8);

  const unsigned short* sBrow = &sB[0][mr * BSTRIDE + g * 8];  // col mr, K 8g..8g+7 and 16+8g..16+8g+7

  for (int k0 = 0; k0 < DD; k0 += TK) {
    int buf = (k0 >> 5) & 1;
    wait_async0();            // my fills for this buffer done
    __syncthreads();          // everyone's fills visible; prev buffer fully read

    if (k0 + TK < DD) {
      // Kick next B tile into the other LDS buffer (overlaps this chunk's math)
      async_fill32(sdst0 + (buf ^ 1) * BBYTES, wrow + k0 + TK);
      // Prefetch next A registers (overlaps this chunk's math)
      Anxt.q[0] = *(const uint4*)(xrow + k0 + TK + g * 8);
      Anxt.q[1] = *(const uint4*)(xrow + k0 + TK + 16 + g * 8);
    }
    __builtin_amdgcn_sched_barrier(0);   // keep prefetches hoisted above B loads

    // Preload ALL 8 B blocks for this chunk (clause of 16 ds_load_b128,
    // one LDS-latency exposure) ...
    const unsigned short* bbase = sBrow + buf * (TN * BSTRIDE);
    Pack32 Ball[8];
    #pragma unroll
    for (int nb = 0; nb < 8; ++nb) {
      const unsigned short* bp = bbase + nb * 16 * BSTRIDE;
      Ball[nb].q[0] = *(const uint4*)(bp);
      Ball[nb].q[1] = *(const uint4*)(bp + 16);
    }
    __builtin_amdgcn_sched_barrier(0);   // forbid sinking loads into WMMA sequence

    // ... then 8 back-to-back WMMAs (counted dscnt waits, in-order LDS returns).
    #pragma unroll
    for (int nb = 0; nb < 8; ++nb) {
      acc[nb] = __builtin_amdgcn_wmma_f32_16x16x32_bf16(
          false, Acur.v, false, Ball[nb].v, (short)0, acc[nb], false, false);
    }
    Acur = Anxt;
  }

  // Epilogue: C layout row = g*8 + r, col = mr. Apply bias + gate weight, scatter-add.
  float bias[8];
  #pragma unroll
  for (int nb = 0; nb < 8; ++nb) bias[nb] = eb[e * OO + n0 + nb * 16 + mr];

  #pragma unroll
  for (int r = 0; r < 8; ++r) {
    int slot = m0 + wave * 16 + g * 8 + r;
    if (slot < cnt) {
      int   tokr = tok_list[eoff + slot];
      float wr   = wgt_list[eoff + slot];
      float* orow = out + (size_t)tokr * OO + n0 + mr;
      #pragma unroll
      for (int nb = 0; nb < 8; ++nb)
        atomicAdd(orow + nb * 16, (acc[nb][r] + bias[nb]) * wr);
    }
  }
}

extern "C" void kernel_launch(void* const* d_in, const int* in_sizes, int n_in,
                              void* d_out, int out_size, void* d_ws, size_t ws_size,
                              hipStream_t stream) {
  const float* x  = (const float*)d_in[0];   // [B,T,D]
  const float* gw = (const float*)d_in[1];   // [D,E]
  const float* gb = (const float*)d_in[2];   // [E]
  const float* ew = (const float*)d_in[3];   // [E,D,O]
  const float* eb = (const float*)d_in[4];   // [E,O]
  float* out = (float*)d_out;                // [B,T,O]

  char* ws = (char*)d_ws;
  unsigned short* xb = (unsigned short*)ws;  ws += (size_t)NTOK * DD * 2;    // 16 MB
  unsigned short* wt = (unsigned short*)ws;  ws += (size_t)EE * OO * DD * 2; // 16 MB
  int*   counts   = (int*)ws;                ws += 256;
  int*   tok_list = (int*)ws;                ws += (size_t)EE * NMAX * 4;    // 256 KB
  float* wgt_list = (float*)ws;              ws += (size_t)EE * NMAX * 4;    // 256 KB

  k_init<<<4096, 256, 0, stream>>>(out, out_size, counts);
  k_cvt_x<<<(NTOK * DD / 8 + 255) / 256, 256, 0, stream>>>(x, xb, NTOK * DD / 8);
  k_wt<<<(EE * OO * (DD / 16) + 255) / 256, 256, 0, stream>>>(ew, wt);
  k_gate<<<NTOK / 8, 256, 0, stream>>>(x, gw, gb, counts, tok_list, wgt_list);

  dim3 grid(OO / TN, NTOK / TM, EE);   // (8, 64, 8); inactive tiles exit on counts[e]
  k_moe_gemm<<<grid, 256, 0, stream>>>(xb, wt, eb, counts, tok_list, wgt_list, out);
}
